// CBMGINInference_26087631356379
// MI455X (gfx1250) — matrix operand order repaired and checked by reading, not compiled, not run on hardware
//
#include <hip/hip_runtime.h>
#include <hip/hip_bf16.h>

#define DD       128
#define N_NODES  100000
#define N_EDGES  1600000
#define M_TILES  (N_NODES / 16)   // 6250, exact
#define WT_STRIDE 136             // padded bf16 row stride (bank-conflict avoidance, keeps 16B align)

typedef __attribute__((ext_vector_type(16))) __bf16 v16bf;
typedef __attribute__((ext_vector_type(8)))  float  v8f;

// ---------------------------------------------------------------------------
// Kernel 1: y = (1 + eps) * x        (float4 streaming)
// ---------------------------------------------------------------------------
__global__ __launch_bounds__(256) void gin_init_y(const float* __restrict__ x,
                                                  const float* __restrict__ eps,
                                                  float* __restrict__ y) {
    const float s = 1.0f + eps[0];
    size_t i = (size_t)blockIdx.x * 256 + threadIdx.x;           // float4 index
    if (i < (size_t)N_NODES * (DD / 4)) {
        const float4* x4 = (const float4*)x;
        float4* y4 = (float4*)y;
        float4 v = x4[i];
        v.x *= s; v.y *= s; v.z *= s; v.w *= s;
        y4[i] = v;
    }
}

// ---------------------------------------------------------------------------
// Kernel 2: scatter-add  y[dst] += x[src]   (one wave32 per edge, float4/lane)
// ---------------------------------------------------------------------------
__global__ __launch_bounds__(256) void gin_scatter(const float* __restrict__ x,
                                                   const int* __restrict__ ei,
                                                   float* __restrict__ y) {
    const int e = blockIdx.x * 8 + (threadIdx.x >> 5);
    if (e >= N_EDGES) return;
    const int lane = threadIdx.x & 31;
    const int src = ei[e];
    const int dst = ei[N_EDGES + e];
    const float4 v = ((const float4*)(x + (size_t)src * DD))[lane];
    float* yd = y + (size_t)dst * DD + lane * 4;
    unsafeAtomicAdd(yd + 0, v.x);   // hardware global_atomic_add_f32
    unsafeAtomicAdd(yd + 1, v.y);
    unsafeAtomicAdd(yd + 2, v.z);
    unsafeAtomicAdd(yd + 3, v.w);
}

// ---------------------------------------------------------------------------
// Kernel 3: fused MLP   out = relu(y@W1 + b1)@W2 + b2
// 256 threads = 8 waves; each wave owns one 16-row M-tile, full 128 columns.
// bf16 WMMA 16x16x32; weights transposed+converted into LDS; h stays in LDS.
// ---------------------------------------------------------------------------
__global__ __launch_bounds__(256) void gin_mlp(const float* __restrict__ y,
                                               const float* __restrict__ W1,
                                               const float* __restrict__ b1,
                                               const float* __restrict__ W2,
                                               const float* __restrict__ b2,
                                               float* __restrict__ out) {
    __shared__ __bf16 wt[DD * WT_STRIDE];            // W^T tile [n][k], bf16, padded
    __shared__ __bf16 hbuf[8 * 16 * WT_STRIDE];      // per-wave h tiles [m][k], bf16

    const int tid  = threadIdx.x;
    const int wave = tid >> 5;
    const int lane = tid & 31;
    const int lmod = lane & 15;
    const int khalf = (lane < 16) ? 0 : 8;           // A-frag K sub-offset
    const int bhalf = (lane < 16) ? 0 : 16;          // unused; B-frag K sub-offset is khalf

    // ---- stage W1^T (fp32 -> bf16); coalesced row-major reads, LDS scatter ----
    for (int i = tid; i < DD * DD; i += 256) {
        const int k = i >> 7, n = i & (DD - 1);      // read W1[k][n] contiguous
        wt[n * WT_STRIDE + k] = (__bf16)W1[i];
    }
    __syncthreads();

    int tile = blockIdx.x * 8 + wave;
    const bool valid = (tile < M_TILES);
    if (!valid) tile = M_TILES - 1;                  // clamp: keep EXEC/barriers uniform
    const int m0  = tile * 16;
    const int row = m0 + lmod;

    // ================= GEMM 1: h = relu(y @ W1 + b1) =================
    v8f acc[8];
#pragma unroll
    for (int nt = 0; nt < 8; ++nt) acc[nt] = (v8f)0.0f;

#pragma unroll
    for (int kf = 0; kf < 4; ++kf) {
        // A fragment straight from global y (fp32 -> bf16), wide b128 loads
        const float* yp = y + (size_t)row * DD + kf * 32 + khalf;
        const float4 f0 = *(const float4*)(yp);
        const float4 f1 = *(const float4*)(yp + 4);
        const float4 f2 = *(const float4*)(yp + 16);
        const float4 f3 = *(const float4*)(yp + 20);
        v16bf a;
        a[0]  = (__bf16)f0.x; a[1]  = (__bf16)f0.y; a[2]  = (__bf16)f0.z; a[3]  = (__bf16)f0.w;
        a[4]  = (__bf16)f1.x; a[5]  = (__bf16)f1.y; a[6]  = (__bf16)f1.z; a[7]  = (__bf16)f1.w;
        a[8]  = (__bf16)f2.x; a[9]  = (__bf16)f2.y; a[10] = (__bf16)f2.z; a[11] = (__bf16)f2.w;
        a[12] = (__bf16)f3.x; a[13] = (__bf16)f3.y; a[14] = (__bf16)f3.z; a[15] = (__bf16)f3.w;
#pragma unroll
        for (int nt = 0; nt < 8; ++nt) {
            const __bf16* bp = &wt[(nt * 16 + lmod) * WT_STRIDE + kf * 32 + khalf];
            v16bf b;
#pragma unroll
            for (int e = 0; e < 16; ++e) b[e] = bp[(e & 7) + ((e >> 3) << 4)];
            acc[nt] = __builtin_amdgcn_wmma_f32_16x16x32_bf16(
                false, a, false, b, (short)0, acc[nt], false, false);
        }
    }

    // bias + relu, spill h tile to LDS (C-layout -> row-major bf16)
    __bf16* hb = &hbuf[wave * 16 * WT_STRIDE];
#pragma unroll
    for (int nt = 0; nt < 8; ++nt) {
        const float bv = b1[nt * 16 + lmod];
#pragma unroll
        for (int r = 0; r < 8; ++r) {
            float v = acc[nt][r] + bv;
            v = v > 0.0f ? v : 0.0f;
            const int m = r + khalf;                 // M = r (lanes<16) or r+8
            hb[m * WT_STRIDE + nt * 16 + lmod] = (__bf16)v;
        }
    }
    __syncthreads();

    // ---- restage W2^T over the same LDS (coalesced) ----
    for (int i = tid; i < DD * DD; i += 256) {
        const int k = i >> 7, n = i & (DD - 1);
        wt[n * WT_STRIDE + k] = (__bf16)W2[i];
    }
    __syncthreads();

    // ================= GEMM 2: out = h @ W2 + b2 =================
#pragma unroll
    for (int nt = 0; nt < 8; ++nt) acc[nt] = (v8f)0.0f;

#pragma unroll
    for (int kf = 0; kf < 4; ++kf) {
        const __bf16* ap = &hb[lmod * WT_STRIDE + kf * 32 + khalf];
        v16bf a;
#pragma unroll
        for (int e = 0; e < 8; ++e) a[e]     = ap[e];
#pragma unroll
        for (int e = 0; e < 8; ++e) a[8 + e] = ap[16 + e];
#pragma unroll
        for (int nt = 0; nt < 8; ++nt) {
            const __bf16* bp = &wt[(nt * 16 + lmod) * WT_STRIDE + kf * 32 + khalf];
            v16bf b;
#pragma unroll
            for (int e = 0; e < 16; ++e) b[e] = bp[(e & 7) + ((e >> 3) << 4)];
            acc[nt] = __builtin_amdgcn_wmma_f32_16x16x32_bf16(
                false, a, false, b, (short)0, acc[nt], false, false);
        }
    }

    if (valid) {
#pragma unroll
        for (int nt = 0; nt < 8; ++nt) {
            const float bv = b2[nt * 16 + lmod];
#pragma unroll
            for (int r = 0; r < 8; ++r) {
                const int m = m0 + r + khalf;
                out[(size_t)m * DD + nt * 16 + lmod] = acc[nt][r] + bv;
            }
        }
    }
}

// ---------------------------------------------------------------------------
extern "C" void kernel_launch(void* const* d_in, const int* in_sizes, int n_in,
                              void* d_out, int out_size, void* d_ws, size_t ws_size,
                              hipStream_t stream) {
    const float* x   = (const float*)d_in[0];
    const int*   ei  = (const int*)d_in[1];   // edge_index [2, N_EDGES] (int32)
    const float* eps = (const float*)d_in[2];
    const float* W1  = (const float*)d_in[3];
    const float* b1  = (const float*)d_in[4];
    const float* W2  = (const float*)d_in[5];
    const float* b2  = (const float*)d_in[6];
    float* out = (float*)d_out;
    float* y   = (float*)d_ws;                // [N_NODES, 128] fp32 scratch (51.2 MB)

    // y = (1+eps)*x
    const int n4 = N_NODES * (DD / 4);
    gin_init_y<<<dim3((n4 + 255) / 256), dim3(256), 0, stream>>>(x, eps, y);

    // y[dst] += x[src]  (8 edges per 256-thread block)
    gin_scatter<<<dim3((N_EDGES + 7) / 8), dim3(256), 0, stream>>>(x, ei, y);

    // fused WMMA MLP: 8 M-tiles (128 rows) per block
    gin_mlp<<<dim3((M_TILES + 7) / 8), dim3(256), 0, stream>>>(y, W1, b1, W2, b2, out);
}
